// LOD_Multi_28484223107123
// MI455X (gfx1250) — matrix operand order repaired and checked by reading, not compiled, not run on hardware
//
#include <hip/hip_runtime.h>

typedef __attribute__((ext_vector_type(16))) _Float16 v16h;
typedef __attribute__((ext_vector_type(8)))  float    v8f;
typedef _Float16 f16;

// ---------------- dims ----------------
#define B_   16
#define S_   256
#define T0_  64
#define NT_  256
#define NE_  128
#define L_   6
#define H_   8
#define HD_  32
#define DFF_ 512
#define ROWS_ (B_*NT_)   // 4096 token rows

__device__ __forceinline__ float gelu_f(float x) {
    return 0.5f * x * (1.0f + erff(x * 0.70710678118654752f));
}

// ---- CDNA5 async global->LDS copy (16B per lane), ASYNCcnt tracked ----
__device__ __forceinline__ void async_cp16(const f16* gptr, f16* sptr) {
    unsigned lds = (unsigned)(unsigned long long)(uintptr_t)sptr; // low 32b of flat = LDS offset
    asm volatile("global_load_async_to_lds_b128 %0, %1, off"
                 :: "v"(lds), "v"((unsigned long long)(uintptr_t)gptr)
                 : "memory");
}
__device__ __forceinline__ void wait_async_le4() {
    asm volatile("s_wait_asynccnt 0x4" ::: "memory");
}
__device__ __forceinline__ void wait_async_le0() {
    asm volatile("s_wait_asynccnt 0x0" ::: "memory");
}

// =====================================================================
// WMMA GEMM, all-f16 operands, async double-buffered LDS staging.
//   A : [M,K] f16 row-major        (lda)
//   Bt: [N,K] f16 (pre-transposed) (ldbt)
//   out: f32 C or f16 Ch = act(A@B + bias)
// Block = 128 threads (4 waves), 64x64 tile, BK=32, 2x2 WMMA per wave.
// =====================================================================
template<bool ACT_GELU, bool OUTF16>
__global__ __launch_bounds__(128)
void gemm_wmma(const f16* __restrict__ A, const f16* __restrict__ Bt,
               const float* __restrict__ bias,
               float* __restrict__ C, f16* __restrict__ Ch,
               int M, int N, int K, int lda, int ldbt, int ldc)
{
    __shared__ __align__(16) f16 As[2][64 * 40];   // [row][k], stride 40 (80B, 16B-aligned)
    __shared__ __align__(16) f16 Bs[2][64 * 40];   // [n][k]

    const int tid  = threadIdx.x;
    const int wave = tid >> 5;
    const int lane = tid & 31;
    const int m0 = blockIdx.y * 64;
    const int n0 = blockIdx.x * 64;
    const int wm = (wave >> 1) * 32;
    const int wn = (wave & 1) * 32;

    const int nk = K >> 5;

    // stage tile kt into buffer s: 64 rows x 32 f16 = 256 16B-chunks, 2 per thread
    auto stage = [&](int kt, int s) {
        const int k0 = kt << 5;
        #pragma unroll
        for (int i = tid; i < 256; i += 128) {
            int r = i >> 2, c = (i & 3) << 3;   // c in halves
            async_cp16(&A[(long)(m0 + r) * lda + k0 + c], &As[s][r * 40 + c]);
        }
        #pragma unroll
        for (int i = tid; i < 256; i += 128) {
            int r = i >> 2, c = (i & 3) << 3;
            async_cp16(&Bt[(long)(n0 + r) * ldbt + k0 + c], &Bs[s][r * 40 + c]);
        }
    };

    v8f acc[2][2] = {};
    const int mr   = lane & 15;
    const int klo  = (lane >> 4) * 8;
    const int koff = (lane >> 4) * 16;

    stage(0, 0);
    for (int kt = 0; kt < nk; ++kt) {
        const int cb = kt & 1;
        if (kt + 1 < nk) { stage(kt + 1, cb ^ 1); wait_async_le4(); }
        else             { wait_async_le0(); }
        __syncthreads();

        v16h af[2], bf[2];
        #pragma unroll
        for (int t = 0; t < 2; ++t) {
            int row = wm + t * 16 + mr;
            #pragma unroll
            for (int j = 0; j < 8; ++j) {
                af[t][j]     = As[cb][row * 40 + klo + j];
                af[t][8 + j] = As[cb][row * 40 + klo + 16 + j];
            }
        }
        #pragma unroll
        for (int t = 0; t < 2; ++t) {
            int col = wn + t * 16 + (lane & 15);
            #pragma unroll
            for (int j = 0; j < 16; ++j)
                bf[t][j] = Bs[cb][col * 40 + klo + (j & 7) + ((j >> 3) << 4)];
        }
        #pragma unroll
        for (int i = 0; i < 2; ++i)
            #pragma unroll
            for (int j = 0; j < 2; ++j)
                acc[i][j] = __builtin_amdgcn_wmma_f32_16x16x32_f16(
                    false, af[i], false, bf[j], (short)0, acc[i][j], false, false);
        __syncthreads();
    }

    // epilogue: C layout -> VGPR r: M = r + 8*(lane>=16), N = lane&15
    const int rbase = (lane >> 4) * 8;
    const int cl    = lane & 15;
    #pragma unroll
    for (int i = 0; i < 2; ++i)
        #pragma unroll
        for (int j = 0; j < 2; ++j) {
            int colg = n0 + wn + j * 16 + cl;
            float bv = bias[colg];
            #pragma unroll
            for (int r = 0; r < 8; ++r) {
                int rowg = m0 + wm + i * 16 + rbase + r;
                float v = acc[i][j][r] + bv;
                if (ACT_GELU) v = gelu_f(v);
                if constexpr (OUTF16) Ch[(long)rowg * ldc + colg] = (f16)v;
                else                  C [(long)rowg * ldc + colg] = v;
            }
        }
}

// =====================================================================
// Fused attention, one block per (b, h, 64-query-row chunk).
// 128 threads = 4 waves; scores in accumulator VGPRs, softmax via
// half-wave shuffles; P staged f16 in LDS; P@V with WMMA. f16 output.
// =====================================================================
__global__ __launch_bounds__(128)
void attn_kernel(const f16* __restrict__ qh, const f16* __restrict__ kh,
                 const f16* __restrict__ vh, f16* __restrict__ outp)
{
    __shared__ f16 Pb[64 * 264];           // ~33 KB

    const int c  = blockIdx.x & 3;
    const int h  = (blockIdx.x >> 2) & 7;
    const int b  = blockIdx.x >> 5;
    const int q0 = c * 64;

    const int tid  = threadIdx.x;
    const int wave = tid >> 5;
    const int lane = tid & 31;
    const int mr   = lane & 15;
    const int klo  = (lane >> 4) * 8;
    const int koff = (lane >> 4) * 16;
    const int rbase = (lane >> 4) * 8;
    const float scale = 0.17677669529663687f;   // 1/sqrt(32)

    const long qrow = (long)(b * NT_ + q0 + wave * 16 + mr) * S_ + h * HD_;
    v16h aq;
    #pragma unroll
    for (int j = 0; j < 8; ++j) {
        aq[j]     = qh[qrow + klo + j];
        aq[8 + j] = qh[qrow + klo + 16 + j];
    }

    v8f sacc[16] = {};
    #pragma unroll
    for (int t = 0; t < 16; ++t) {
        const long krow = (long)(b * NT_ + t * 16 + mr) * S_ + h * HD_;
        v16h bk;
        #pragma unroll
        for (int j = 0; j < 16; ++j) bk[j] = kh[krow + klo + (j & 7) + ((j >> 3) << 4)];
        sacc[t] = __builtin_amdgcn_wmma_f32_16x16x32_f16(
            false, aq, false, bk, (short)0, sacc[t], false, false);
    }

    #pragma unroll
    for (int q = 0; q < 8; ++q) {
        float m = -3.4e38f;
        #pragma unroll
        for (int t = 0; t < 16; ++t) m = fmaxf(m, sacc[t][q]);
        #pragma unroll
        for (int o = 1; o < 16; o <<= 1) m = fmaxf(m, __shfl_xor(m, o, 32));
        m *= scale;
        float sum = 0.f;
        #pragma unroll
        for (int t = 0; t < 16; ++t) {
            float e = __expf(sacc[t][q] * scale - m);
            sacc[t][q] = e; sum += e;
        }
        #pragma unroll
        for (int o = 1; o < 16; o <<= 1) sum += __shfl_xor(sum, o, 32);
        float inv = 1.0f / sum;
        int prow = wave * 16 + q + rbase;
        #pragma unroll
        for (int t = 0; t < 16; ++t)
            Pb[prow * 264 + t * 16 + (lane & 15)] = (f16)(sacc[t][q] * inv);
    }
    __syncthreads();

    v8f oacc[2] = {};
    const int arow = wave * 16 + mr;
    #pragma unroll
    for (int k0 = 0; k0 < 256; k0 += 32) {
        v16h ap;
        #pragma unroll
        for (int j = 0; j < 8; ++j) {
            ap[j]     = Pb[arow * 264 + k0 + klo + j];
            ap[8 + j] = Pb[arow * 264 + k0 + klo + 16 + j];
        }
        #pragma unroll
        for (int ct = 0; ct < 2; ++ct) {
            v16h bv;
            #pragma unroll
            for (int j = 0; j < 16; ++j)
                bv[j] = vh[(long)(b * NT_ + k0 + klo + (j & 7) + ((j >> 3) << 4)) * S_ + h * HD_ + ct * 16 + (lane & 15)];
            oacc[ct] = __builtin_amdgcn_wmma_f32_16x16x32_f16(
                false, ap, false, bv, (short)0, oacc[ct], false, false);
        }
    }
    #pragma unroll
    for (int ct = 0; ct < 2; ++ct)
        #pragma unroll
        for (int r = 0; r < 8; ++r) {
            int rowg = b * NT_ + q0 + wave * 16 + rbase + r;
            outp[(long)rowg * S_ + h * HD_ + ct * 16 + (lane & 15)] = (f16)oacc[ct][r];
        }
}

// ---------------- residual + LayerNorm; writes f32 and f16 copies ----------------
__global__ __launch_bounds__(256)
void add_ln_kernel(const float* __restrict__ x, const float* __restrict__ a,
                   const float* __restrict__ g, const float* __restrict__ be,
                   float* __restrict__ outp, f16* __restrict__ outph)
{
    __shared__ float red[8];
    const int row = blockIdx.x, tid = threadIdx.x;
    float v = x[(long)row * 256 + tid] + a[(long)row * 256 + tid];

    float s = v;
    #pragma unroll
    for (int o = 16; o > 0; o >>= 1) s += __shfl_xor(s, o, 32);
    if ((tid & 31) == 0) red[tid >> 5] = s;
    __syncthreads();
    float mean = 0.f;
    #pragma unroll
    for (int i = 0; i < 8; ++i) mean += red[i];
    mean *= (1.0f / 256.0f);
    __syncthreads();

    float d = v - mean;
    float q = d * d;
    #pragma unroll
    for (int o = 16; o > 0; o >>= 1) q += __shfl_xor(q, o, 32);
    if ((tid & 31) == 0) red[tid >> 5] = q;
    __syncthreads();
    float var = 0.f;
    #pragma unroll
    for (int i = 0; i < 8; ++i) var += red[i];
    var *= (1.0f / 256.0f);

    float o_ = d * rsqrtf(var + 1e-5f) * g[tid] + be[tid];
    outp [(long)row * 256 + tid] = o_;
    outph[(long)row * 256 + tid] = (f16)o_;
}

// ---- transpose [B,S,NT] -> [B,NT,S], add pe[s][t]; X/cur in f32+f16 ----
__global__ void transpose_add_pe(const float* __restrict__ txin, const float* __restrict__ pe,
                                 float* __restrict__ X, f16* __restrict__ Xh,
                                 float* __restrict__ cur, f16* __restrict__ curh)
{
    __shared__ float tile[16][17];
    const int b = blockIdx.z;
    const int t0 = blockIdx.x * 16, s0 = blockIdx.y * 16;
    const int tx = threadIdx.x, ty = threadIdx.y;
    int s = s0 + ty, t = t0 + tx;
    tile[ty][tx] = txin[(long)(b * S_ + s) * NT_ + t] + pe[s * NT_ + t];
    __syncthreads();
    int tt = t0 + ty, ss = s0 + tx;
    float o = tile[tx][ty];
    long oi = (long)(b * NT_ + tt) * S_ + ss;
    X[oi] = o; cur[oi] = o;
    Xh[oi] = (f16)o; curh[oi] = (f16)o;
}

// ---- x_emb = cur + X, f16 output for the head GEMM ----
__global__ void add_res_kernel(const float* __restrict__ cur, const float* __restrict__ X,
                               f16* __restrict__ outh)
{
    long i = (long)blockIdx.x * blockDim.x + threadIdx.x;
    outh[i] = (f16)(cur[i] + X[i]);
}

// ---- weight prep: f32 [R,C] -> f16 transposed [C,R] ----
__global__ void tconv_kernel(const float* __restrict__ in, f16* __restrict__ out,
                             int R, int C)
{
    __shared__ float t[16][17];
    const int c0 = blockIdx.x * 16, r0 = blockIdx.y * 16;
    const int tx = threadIdx.x, ty = threadIdx.y;
    t[ty][tx] = in[(long)(r0 + ty) * C + c0 + tx];
    __syncthreads();
    out[(long)(c0 + ty) * R + r0 + tx] = (f16)t[tx][ty];
}

// ---- plain f32 -> f16 convert ----
__global__ void conv_kernel(const float* __restrict__ in, f16* __restrict__ out)
{
    long i = (long)blockIdx.x * blockDim.x + threadIdx.x;
    out[i] = (f16)in[i];
}

// ---------------- fused gather + dual einsum over bases ----------------
__global__ __launch_bounds__(64)
void einsum_kernel(const float* __restrict__ pc, const float* __restrict__ lc,
                   const float* __restrict__ bases, const int* __restrict__ bnum,
                   float* __restrict__ pred, float* __restrict__ lat)
{
    __shared__ float spc[128], slc[128];
    const int bt = blockIdx.x;
    const int b = bt >> 8, t = bt & 255;
    const int tid = threadIdx.x;
    for (int i = tid; i < 128; i += 64) {
        spc[i] = pc[(long)bt * 128 + i];
        slc[i] = lc[(long)bt * 128 + i];
    }
    __syncthreads();
    const long base = ((long)bnum[b] * NT_ + t) * (long)(NE_ * S_);
    const float4* G = (const float4*)(bases + base);
    float p0=0,p1=0,p2=0,p3=0, l0=0,l1=0,l2=0,l3=0;
    for (int n = 0; n < 128; ++n) {
        float4 g = G[n * 64 + tid];
        float a = spc[n], cl_ = slc[n];
        p0 += a * g.x; p1 += a * g.y; p2 += a * g.z; p3 += a * g.w;
        l0 += cl_ * g.x; l1 += cl_ * g.y; l2 += cl_ * g.z; l3 += cl_ * g.w;
    }
    ((float4*)(pred + (long)bt * 256))[tid] = make_float4(p0, p1, p2, p3);
    ((float4*)(lat  + (long)bt * 256))[tid] = make_float4(l0, l1, l2, l3);
}

// =====================================================================
static void run_gemm(hipStream_t s, const f16* A, const f16* Bt, const float* bias,
                     float* C, f16* Ch, int M, int N, int K,
                     int lda, int ldbt, int ldc, bool act, bool outf16)
{
    dim3 grid(N / 64, M / 64), blk(128);
    if (outf16) {
        if (act) gemm_wmma<true,  true ><<<grid, blk, 0, s>>>(A, Bt, bias, C, Ch, M, N, K, lda, ldbt, ldc);
        else     gemm_wmma<false, true ><<<grid, blk, 0, s>>>(A, Bt, bias, C, Ch, M, N, K, lda, ldbt, ldc);
    } else {
        if (act) gemm_wmma<true,  false><<<grid, blk, 0, s>>>(A, Bt, bias, C, Ch, M, N, K, lda, ldbt, ldc);
        else     gemm_wmma<false, false><<<grid, blk, 0, s>>>(A, Bt, bias, C, Ch, M, N, K, lda, ldbt, ldc);
    }
}

static void run_tconv(hipStream_t s, const float* in, f16* out, int R, int C)
{
    tconv_kernel<<<dim3(C / 16, R / 16), dim3(16, 16), 0, s>>>(in, out, R, C);
}

extern "C" void kernel_launch(void* const* d_in, const int* in_sizes, int n_in,
                              void* d_out, int out_size, void* d_ws, size_t ws_size,
                              hipStream_t stream)
{
    const float* seq    = (const float*)d_in[0];
    const int*   bnum   = (const int*)  d_in[1];
    const float* tp_w1  = (const float*)d_in[2];
    const float* tp_b1  = (const float*)d_in[3];
    const float* tp_w2  = (const float*)d_in[4];
    const float* tp_b2  = (const float*)d_in[5];
    const float* pos_enc= (const float*)d_in[6];
    const float* pe_w1  = (const float*)d_in[7];
    const float* pe_b1  = (const float*)d_in[8];
    const float* pe_w2  = (const float*)d_in[9];
    const float* pe_b2  = (const float*)d_in[10];
    const float* wq = (const float*)d_in[11]; const float* bq = (const float*)d_in[12];
    const float* wk = (const float*)d_in[13]; const float* bk = (const float*)d_in[14];
    const float* wv = (const float*)d_in[15]; const float* bv = (const float*)d_in[16];
    const float* wo = (const float*)d_in[17]; const float* bo = (const float*)d_in[18];
    const float* w1 = (const float*)d_in[19]; const float* b1 = (const float*)d_in[20];
    const float* w2 = (const float*)d_in[21]; const float* b2 = (const float*)d_in[22];
    const float* ln1g = (const float*)d_in[23]; const float* ln1b = (const float*)d_in[24];
    const float* ln2g = (const float*)d_in[25]; const float* ln2b = (const float*)d_in[26];
    const float* o_w1 = (const float*)d_in[27]; const float* o_b1 = (const float*)d_in[28];
    const float* o_w2 = (const float*)d_in[29]; const float* o_b2 = (const float*)d_in[30];
    const float* l_w1 = (const float*)d_in[31]; const float* l_b1 = (const float*)d_in[32];
    const float* l_w2 = (const float*)d_in[33]; const float* l_b2 = (const float*)d_in[34];
    const float* bases= (const float*)d_in[35];

    // ---- workspace carve ----
    char* w = (char*)d_ws; size_t off = 0;
    auto carve = [&](size_t bytes) { void* p = w + off; off = (off + bytes + 255) & ~(size_t)255; return p; };
    // f16 transposed weights [N,K]
    f16* posT  = (f16*)carve((size_t)S_ * T0_ * 2);         // [256,64]
    f16* seqh  = (f16*)carve((size_t)ROWS_ * T0_ * 2);      // [4096,64]
    f16* tpw1T = (f16*)carve((size_t)NT_ * T0_ * 2);
    f16* tpw2T = (f16*)carve((size_t)NT_ * NT_ * 2);
    f16* pew1T = (f16*)carve((size_t)NT_ * T0_ * 2);
    f16* pew2T = (f16*)carve((size_t)NT_ * NT_ * 2);
    f16* lw1T  = (f16*)carve((size_t)NE_ * S_ * 2);
    f16* lw2T  = (f16*)carve((size_t)NE_ * NE_ * 2);
    f16* ow1T  = (f16*)carve((size_t)NE_ * S_ * 2);
    f16* ow2T  = (f16*)carve((size_t)NE_ * NE_ * 2);
    f16* wqT   = (f16*)carve((size_t)L_ * S_ * S_ * 2);
    f16* wkT   = (f16*)carve((size_t)L_ * S_ * S_ * 2);
    f16* wvT   = (f16*)carve((size_t)L_ * S_ * S_ * 2);
    f16* woT   = (f16*)carve((size_t)L_ * S_ * S_ * 2);
    f16* w1T   = (f16*)carve((size_t)L_ * S_ * DFF_ * 2);
    f16* w2T   = (f16*)carve((size_t)L_ * DFF_ * S_ * 2);
    // activations
    float* peB  = (float*)carve((size_t)S_ * NT_ * 4);      // pe [S,NT] f32
    f16*   bufAh= (f16*)  carve((size_t)ROWS_ * DFF_ * 2);  // f16 hiddens
    float* tx   = (float*)carve((size_t)ROWS_ * NT_ * 4);
    float* X    = (float*)carve((size_t)ROWS_ * S_ * 4);
    f16*   Xh   = (f16*)  carve((size_t)ROWS_ * S_ * 2);
    float* cur  = (float*)carve((size_t)ROWS_ * S_ * 4);
    f16*   curh = (f16*)  carve((size_t)ROWS_ * S_ * 2);
    f16*   attnh= (f16*)  carve((size_t)ROWS_ * S_ * 2);
    float* tmp  = (float*)carve((size_t)ROWS_ * S_ * 4);
    f16*   qh   = (f16*)  carve((size_t)ROWS_ * S_ * 2);
    f16*   kh   = (f16*)  carve((size_t)ROWS_ * S_ * 2);
    f16*   vh   = (f16*)  carve((size_t)ROWS_ * S_ * 2);

    float* out_pred = (float*)d_out;
    float* out_lat  = out_pred + (size_t)B_ * NT_ * S_;
    float* out_pc   = out_lat  + (size_t)B_ * NT_ * S_;
    float* out_lc   = out_pc   + (size_t)B_ * NT_ * NE_;

    // ---- one-time f16 weight prep (transposed to [N,K]) ----
    run_tconv(stream, pos_enc, posT,  T0_, S_);   // A for pe-MLP: [S,T0]
    run_tconv(stream, tp_w1,   tpw1T, T0_, NT_);
    run_tconv(stream, tp_w2,   tpw2T, NT_, NT_);
    run_tconv(stream, pe_w1,   pew1T, T0_, NT_);
    run_tconv(stream, pe_w2,   pew2T, NT_, NT_);
    run_tconv(stream, l_w1,    lw1T,  S_,  NE_);
    run_tconv(stream, l_w2,    lw2T,  NE_, NE_);
    run_tconv(stream, o_w1,    ow1T,  S_,  NE_);
    run_tconv(stream, o_w2,    ow2T,  NE_, NE_);
    for (int l = 0; l < L_; ++l) {
        run_tconv(stream, wq + (size_t)l*S_*S_,   wqT + (size_t)l*S_*S_,   S_,   S_);
        run_tconv(stream, wk + (size_t)l*S_*S_,   wkT + (size_t)l*S_*S_,   S_,   S_);
        run_tconv(stream, wv + (size_t)l*S_*S_,   wvT + (size_t)l*S_*S_,   S_,   S_);
        run_tconv(stream, wo + (size_t)l*S_*S_,   woT + (size_t)l*S_*S_,   S_,   S_);
        run_tconv(stream, w1 + (size_t)l*S_*DFF_, w1T + (size_t)l*S_*DFF_, S_,   DFF_);
        run_tconv(stream, w2 + (size_t)l*DFF_*S_, w2T + (size_t)l*DFF_*S_, DFF_, S_);
    }
    conv_kernel<<<(ROWS_ * T0_) / 256, 256, 0, stream>>>(seq, seqh);

    // ---- positional-encoding MLP: pe = mlp2(pos_enc^T) ----
    run_gemm(stream, posT,  pew1T, pe_b1, nullptr, bufAh, S_, NT_, T0_, T0_, T0_, NT_, true,  true);
    run_gemm(stream, bufAh, pew2T, pe_b2, peB,     nullptr, S_, NT_, NT_, NT_, NT_, NT_, false, false);

    // ---- time projection: tx = mlp2(seq)  [B*S, NT] ----
    run_gemm(stream, seqh,  tpw1T, tp_b1, nullptr, bufAh, ROWS_, NT_, T0_, T0_, T0_, NT_, true,  true);
    run_gemm(stream, bufAh, tpw2T, tp_b2, tx,      nullptr, ROWS_, NT_, NT_, NT_, NT_, NT_, false, false);

    // ---- X[b,t,s] = tx[b,s,t] + pe[s,t]; cur = X (f32 + f16) ----
    transpose_add_pe<<<dim3(NT_/16, S_/16, B_), dim3(16,16), 0, stream>>>(tx, peB, X, Xh, cur, curh);

    // ---- latent_coeff = mlp2(X) ----
    run_gemm(stream, Xh,    lw1T, l_b1, nullptr, bufAh, ROWS_, NE_, S_,  S_,  S_,  NE_, true,  true);
    run_gemm(stream, bufAh, lw2T, l_b2, out_lc,  nullptr, ROWS_, NE_, NE_, NE_, NE_, NE_, false, false);

    // ---- 6 encoder layers ----
    for (int l = 0; l < L_; ++l) {
        const f16* wqTl = wqT + (size_t)l * S_ * S_;   const float* bql = bq + (size_t)l * S_;
        const f16* wkTl = wkT + (size_t)l * S_ * S_;   const float* bkl = bk + (size_t)l * S_;
        const f16* wvTl = wvT + (size_t)l * S_ * S_;   const float* bvl = bv + (size_t)l * S_;
        const f16* woTl = woT + (size_t)l * S_ * S_;   const float* bol = bo + (size_t)l * S_;
        const f16* w1Tl = w1T + (size_t)l * S_ * DFF_; const float* b1l = b1 + (size_t)l * DFF_;
        const f16* w2Tl = w2T + (size_t)l * DFF_ * S_; const float* b2l = b2 + (size_t)l * S_;

        run_gemm(stream, curh, wqTl, bql, nullptr, qh, ROWS_, S_, S_, S_, S_, S_, false, true);
        run_gemm(stream, curh, wkTl, bkl, nullptr, kh, ROWS_, S_, S_, S_, S_, S_, false, true);
        run_gemm(stream, curh, wvTl, bvl, nullptr, vh, ROWS_, S_, S_, S_, S_, S_, false, true);

        attn_kernel<<<B_ * H_ * 4, 128, 0, stream>>>(qh, kh, vh, attnh);

        run_gemm(stream, attnh, woTl, bol, tmp, nullptr, ROWS_, S_, S_, S_, S_, S_, false, false);
        add_ln_kernel<<<ROWS_, 256, 0, stream>>>(cur, tmp, ln1g + l * S_, ln1b + l * S_, cur, curh);

        run_gemm(stream, curh,  w1Tl, b1l, nullptr, bufAh, ROWS_, DFF_, S_,   S_,   S_,   DFF_, true,  true);
        run_gemm(stream, bufAh, w2Tl, b2l, tmp,     nullptr, ROWS_, S_,   DFF_, DFF_, DFF_, S_,   false, false);
        add_ln_kernel<<<ROWS_, 256, 0, stream>>>(cur, tmp, ln2g + l * S_, ln2b + l * S_, cur, curh);
    }

    // ---- x_emb = encoder(x) + x (f16 for head GEMM) ----
    add_res_kernel<<<ROWS_, 256, 0, stream>>>(cur, X, curh);

    // ---- pred_coeff = mlp2(x_emb) ----
    run_gemm(stream, curh,  ow1T, o_b1, nullptr, bufAh, ROWS_, NE_, S_,  S_,  S_,  NE_, true,  true);
    run_gemm(stream, bufAh, ow2T, o_b2, out_pc,  nullptr, ROWS_, NE_, NE_, NE_, NE_, NE_, false, false);

    // ---- gather + dual einsum over bases (HBM-bound, 537 MB streamed once) ----
    einsum_kernel<<<B_ * NT_, 64, 0, stream>>>(out_pc, out_lc, bases, bnum, out_pred, out_lat);
}
